// KipfGCN_9947144258272
// MI455X (gfx1250) — matrix operand order repaired and checked by reading, not compiled, not run on hardware
//
#include <hip/hip_runtime.h>

// ---------------------------------------------------------------------------
// 2-layer Kipf GCN, MI455X (gfx1250, wave32, WMMA).
//   out = log_softmax( Â relu( Â (x W1) + b1 ) W2 + b2 )
// Â(h W2) = (Âh) W2 -> aggregate the 64-wide h first, then one
// [8192,64]x[64,8192] bf16-WMMA GEMM fused with row log_softmax so the 256MB
// output is written exactly once (HBM roofline ~11us @ 23.3 TB/s; everything
// else is L2-resident and <2us combined).
// All WMMA operands are pre-converted to bf16 so the GEMM inner loops are
// pure global_load_b128 + v_wmma (no per-iteration cvt VALU).
// ---------------------------------------------------------------------------

#define N_NODES 8192
#define F_IN    512
#define H_DIM   64
#define N_EDGES (N_NODES * 64)

typedef __attribute__((ext_vector_type(16))) __bf16 v16bf;
typedef __attribute__((ext_vector_type(4)))  __bf16 v4bf;
typedef __attribute__((ext_vector_type(8)))  float  v8f;

union BF16Frag { v16bf v; uint4 u[2]; };

// A fragment (16x32 bf16) from row-major bf16 [*, ld].
// Lane L holds row L%16; half-lane kb=(L/16)*16 covers
// K = {k0+kb..+15} in elems 0..15 (matches load_b_bf16).
__device__ __forceinline__ v16bf load_a_bf16(const __bf16* __restrict__ src, int rowBase,
                                             int ld, int k0, int lane) {
  const int r  = rowBase + (lane & 15);
  const int kb = (lane >> 4) * 16;
  const __bf16* p = src + (size_t)r * ld + k0 + kb;
  BF16Frag a;
  a.u[0] = *(const uint4*)p;          // 16B: K = k0+kb .. +7
  a.u[1] = *(const uint4*)(p + 8);    // 16B: K = k0+kb+8 .. +15
  return a.v;
}

// B fragment (32x16 bf16) from pre-transposed bf16 wt[N][ldk] (wt[n][k]=W[k][n]):
// lane L holds col L%16, K = {k0+(L/16)*16 .. +15} -> 32 contiguous bytes.
__device__ __forceinline__ v16bf load_b_bf16(const __bf16* __restrict__ wt, int colBase,
                                             int ldk, int k0, int lane) {
  const __bf16* p = wt + (size_t)(colBase + (lane & 15)) * ldk + k0 + ((lane >> 4) << 4);
  BF16Frag b;
  b.u[0] = ((const uint4*)p)[0];
  b.u[1] = ((const uint4*)p)[1];
  return b.v;
}

// ------------------------------- degree / norm ------------------------------

__global__ void k_deg_init(float* deg) {               // self-loop contributes 1
  int i = blockIdx.x * 256 + threadIdx.x;
  if (i < N_NODES) deg[i] = 1.0f;
}

__global__ void k_deg_edges(const int* __restrict__ ei, float* deg) {
  int e = blockIdx.x * 256 + threadIdx.x;
  if (e < N_EDGES) atomicAdd(&deg[ei[N_EDGES + e]], 1.0f);   // dst
}

__global__ void k_dinv(float* deg) {
  int i = blockIdx.x * 256 + threadIdx.x;
  if (i < N_NODES) { float d = deg[i]; deg[i] = (d > 0.f) ? rsqrtf(d) : 0.f; }
}

// --------------------------- bf16 conversions -------------------------------

__global__ void k_cvt_bf16x4(const float4* __restrict__ in, v4bf* __restrict__ out_, int n4) {
  int i = blockIdx.x * 256 + threadIdx.x;
  if (i < n4) {
    float4 f = in[i];
    v4bf o;
    o[0] = (__bf16)f.x; o[1] = (__bf16)f.y; o[2] = (__bf16)f.z; o[3] = (__bf16)f.w;
    out_[i] = o;
  }
}

__global__ void k_w1t(const float* __restrict__ W1, __bf16* __restrict__ w1t) {
  int i = blockIdx.x * 256 + threadIdx.x;                    // [64][512]
  if (i < H_DIM * F_IN) {
    int j = i >> 9, k = i & 511;
    w1t[i] = (__bf16)W1[k * H_DIM + j];
  }
}

__global__ void k_w2t(const float* __restrict__ W2, __bf16* __restrict__ w2t) {
  int i = blockIdx.x * 256 + threadIdx.x;                    // [8192][64]
  if (i < N_NODES * H_DIM) {
    int j = i >> 6, k = i & 63;
    w2t[i] = (__bf16)W2[(size_t)k * N_NODES + j];
  }
}

// ------------------------ layer-1 GEMM: XW = x @ W1 -------------------------
// 2048 tiles (512 row-tiles x 4 col-tiles); one 16x16 tile per wave, K=512.
// Inner loop: 4x global_load_b128 + 1x v_wmma, fully unrolled (16 steps).

__global__ __launch_bounds__(256) void k_gemm1(const __bf16* __restrict__ xb,
                                               const __bf16* __restrict__ w1t,
                                               float* __restrict__ xw) {
  const int lane = threadIdx.x & 31;
  const int wave = threadIdx.x >> 5;
  const int tile = blockIdx.x * 8 + wave;
  const int rowBase = (tile >> 2) * 16;
  const int colBase = (tile & 3) * 16;

  v8f acc = {0.f, 0.f, 0.f, 0.f, 0.f, 0.f, 0.f, 0.f};
#pragma unroll
  for (int k0 = 0; k0 < F_IN; k0 += 32) {
    v16bf a = load_a_bf16(xb, rowBase, F_IN, k0, lane);
    v16bf b = load_b_bf16(w1t, colBase, F_IN, k0, lane);
    acc = __builtin_amdgcn_wmma_f32_16x16x32_bf16(false, a, false, b, (short)0, acc, false, false);
  }
  const int cg = colBase + (lane & 15);
  const int rh = rowBase + ((lane >> 4) << 3);
#pragma unroll
  for (int v = 0; v < 8; ++v) xw[(size_t)(rh + v) * H_DIM + cg] = acc[v];
}

// ----------------------------- aggregation ----------------------------------

__global__ void k_self_init(const float* __restrict__ in, const float* __restrict__ dinv,
                            float* __restrict__ out) {      // self loop: dinv[i]^2 * in[i,:]
  int i = blockIdx.x * 256 + threadIdx.x;
  if (i < N_NODES * H_DIM) { float dv = dinv[i >> 6]; out[i] = in[i] * dv * dv; }
}

__global__ void k_edge_agg(const int* __restrict__ ei, const float* __restrict__ dinv,
                           const float* __restrict__ in, float* __restrict__ out) {
  int t = blockIdx.x * 256 + threadIdx.x;                   // one (edge, feature) each
  if (t < N_EDGES * H_DIM) {
    int e = t >> 6, f = t & 63;
    int s = ei[e], d = ei[N_EDGES + e];
    float w = dinv[s] * dinv[d];
    atomicAdd(&out[d * H_DIM + f], in[s * H_DIM + f] * w);  // 2MB target -> L2 atomics
  }
}

__global__ void k_relu_bias(float* __restrict__ h, const float* __restrict__ b1) {
  int i = blockIdx.x * 256 + threadIdx.x;
  if (i < N_NODES * H_DIM) h[i] = fmaxf(h[i] + b1[i & 63], 0.f);
}

// ------------- fused layer-2 GEMM + log_softmax (write 256MB once) ----------
// 512 blocks x 256 threads. Block owns 16 rows; wave w owns cols [w*1024, +1024).
// Pass 1: flash-style online (rowmax, sumexp), accumulators discarded.
// Pass 2: recompute tiles (compute is ~free vs the HBM write) and store
//         logit - (m + log s). W2^T (1MB bf16) is L2-resident for all blocks.

__global__ __launch_bounds__(256) void k_final(const __bf16* __restrict__ gb,
                                               const __bf16* __restrict__ w2t,
                                               const float* __restrict__ b2,
                                               float* __restrict__ out) {
  const int lane = threadIdx.x & 31;
  const int wave = threadIdx.x >> 5;
  const int rowBase = blockIdx.x * 16;

  __shared__ float part_m[8][16];
  __shared__ float part_s[8][16];
  __shared__ float rowoff[16];

  // A fragments: 16 rows x K=64 of g (bf16), shared by both passes.
  const v16bf a0 = load_a_bf16(gb, rowBase, H_DIM, 0, lane);
  const v16bf a1 = load_a_bf16(gb, rowBase, H_DIM, 32, lane);

  const int tile0 = wave * 64;

  float m[8], s[8];
#pragma unroll
  for (int v = 0; v < 8; ++v) { m[v] = -3.4e38f; s[v] = 0.f; }

  // ---- pass 1: online max / sum-exp ----
  for (int t = 0; t < 64; ++t) {
    const int col = (tile0 + t) << 4;
    if (t + 1 < 64) __builtin_prefetch(w2t + (size_t)(col + 16) * H_DIM, 0, 3);
    const float bias = b2[col + (lane & 15)];
    v8f acc;
#pragma unroll
    for (int v = 0; v < 8; ++v) acc[v] = bias;
    v16bf b0 = load_b_bf16(w2t, col, H_DIM, 0, lane);
    v16bf b1 = load_b_bf16(w2t, col, H_DIM, 32, lane);
    acc = __builtin_amdgcn_wmma_f32_16x16x32_bf16(false, a0, false, b0, (short)0, acc, false, false);
    acc = __builtin_amdgcn_wmma_f32_16x16x32_bf16(false, a1, false, b1, (short)0, acc, false, false);
#pragma unroll
    for (int v = 0; v < 8; ++v) {
      float x  = acc[v];
      float mn = fmaxf(m[v], x);
      s[v] = s[v] * __expf(m[v] - mn) + __expf(x - mn);
      m[v] = mn;
    }
  }

  // Reduce (m,s) across the 16 lanes of each half-wave (one row per half-lane).
#pragma unroll
  for (int v = 0; v < 8; ++v) {
    float mv = m[v], sv = s[v];
    for (int off = 1; off < 16; off <<= 1) {
      float m2 = __shfl_xor(mv, off, 32);
      float s2 = __shfl_xor(sv, off, 32);
      float mn = fmaxf(mv, m2);
      sv = sv * __expf(mv - mn) + s2 * __expf(m2 - mn);
      mv = mn;
    }
    m[v] = mv; s[v] = sv;
  }
  if ((lane & 15) == 0) {
    const int rhalf = (lane >> 4) << 3;
#pragma unroll
    for (int v = 0; v < 8; ++v) { part_m[wave][rhalf + v] = m[v]; part_s[wave][rhalf + v] = s[v]; }
  }
  __syncthreads();
  if (threadIdx.x < 16) {
    float mv = -3.4e38f, sv = 0.f;
#pragma unroll
    for (int w = 0; w < 8; ++w) {
      float m2 = part_m[w][threadIdx.x], s2 = part_s[w][threadIdx.x];
      float mn = fmaxf(mv, m2);
      sv = sv * __expf(mv - mn) + s2 * __expf(m2 - mn);
      mv = mn;
    }
    rowoff[threadIdx.x] = mv + __logf(sv);
  }
  __syncthreads();

  float roff[8];
#pragma unroll
  for (int v = 0; v < 8; ++v) roff[v] = rowoff[v + ((lane >> 4) << 3)];

  // ---- pass 2: recompute and store normalized logits ----
  for (int t = 0; t < 64; ++t) {
    const int col = (tile0 + t) << 4;
    const float bias = b2[col + (lane & 15)];
    v8f acc;
#pragma unroll
    for (int v = 0; v < 8; ++v) acc[v] = bias;
    v16bf b0 = load_b_bf16(w2t, col, H_DIM, 0, lane);
    v16bf b1 = load_b_bf16(w2t, col, H_DIM, 32, lane);
    acc = __builtin_amdgcn_wmma_f32_16x16x32_bf16(false, a0, false, b0, (short)0, acc, false, false);
    acc = __builtin_amdgcn_wmma_f32_16x16x32_bf16(false, a1, false, b1, (short)0, acc, false, false);
    const int cg = col + (lane & 15);
    const int rh = rowBase + ((lane >> 4) << 3);
#pragma unroll
    for (int v = 0; v < 8; ++v)
      out[(size_t)(rh + v) * N_NODES + cg] = acc[v] - roff[v];
  }
}

// ---------------------------------------------------------------------------

extern "C" void kernel_launch(void* const* d_in, const int* in_sizes, int n_in,
                              void* d_out, int out_size, void* d_ws, size_t ws_size,
                              hipStream_t stream) {
  const float* x  = (const float*)d_in[0];
  const int*   ei = (const int*)d_in[1];     // [2][E]: src row then dst row
  const float* W1 = (const float*)d_in[2];
  const float* b1 = (const float*)d_in[3];
  const float* W2 = (const float*)d_in[4];
  const float* b2 = (const float*)d_in[5];
  float* out = (float*)d_out;

  // Workspace layout (~16.2 MB)
  char* w = (char*)d_ws;
  float*  dinv = (float*)w;                                     // 32 KB
  __bf16* w1t  = (__bf16*)(w + (32 << 10));                     // 64 KB  [64][512]
  __bf16* w2t  = (__bf16*)(w + (96 << 10));                     // 1 MB   [8192][64]
  float*  xw   = (float*)(w + (96 << 10) + (1 << 20));          // 2 MB
  float*  h    = xw + N_NODES * H_DIM;                          // 2 MB
  float*  g    = h  + N_NODES * H_DIM;                          // 2 MB
  __bf16* xb   = (__bf16*)(g + N_NODES * H_DIM);                // 8 MB   [8192][512]
  __bf16* gb   = (__bf16*)((char*)xb + (size_t)N_NODES * F_IN * 2); // 1 MB [8192][64]

  const int T = 256;
  // degrees -> dinv
  k_deg_init <<<N_NODES / T, T, 0, stream>>>(dinv);
  k_deg_edges<<<N_EDGES / T, T, 0, stream>>>(ei, dinv);
  k_dinv     <<<N_NODES / T, T, 0, stream>>>(dinv);
  // operand prep: bf16 copies / transposes
  k_cvt_bf16x4<<<(N_NODES * F_IN / 4) / T, T, 0, stream>>>((const float4*)x, (v4bf*)xb,
                                                           N_NODES * F_IN / 4);
  k_w1t<<<(H_DIM * F_IN) / T, T, 0, stream>>>(W1, w1t);
  k_w2t<<<(N_NODES * H_DIM) / T, T, 0, stream>>>(W2, w2t);
  // layer 1: XW = x @ W1 (WMMA bf16)
  k_gemm1<<<256, T, 0, stream>>>(xb, w1t, xw);
  // h = Â XW ; relu(h + b1)
  k_self_init<<<(N_NODES * H_DIM) / T, T, 0, stream>>>(xw, dinv, h);
  k_edge_agg <<<(N_EDGES * H_DIM) / T, T, 0, stream>>>(ei, dinv, xw, h);
  k_relu_bias<<<(N_NODES * H_DIM) / T, T, 0, stream>>>(h, b1);
  // g = Â h
  k_self_init<<<(N_NODES * H_DIM) / T, T, 0, stream>>>(h, dinv, g);
  k_edge_agg <<<(N_EDGES * H_DIM) / T, T, 0, stream>>>(ei, dinv, h, g);
  k_cvt_bf16x4<<<(N_NODES * H_DIM / 4) / T, T, 0, stream>>>((const float4*)g, (v4bf*)gb,
                                                            N_NODES * H_DIM / 4);
  // fused: log_softmax(g @ W2 + b2), single 256MB write
  k_final<<<N_NODES / 16, T, 0, stream>>>(gb, w2t, b2, out);

  (void)in_sizes; (void)n_in; (void)out_size; (void)ws_size;
}
